// GraphMambaModel_45140106281147
// MI455X (gfx1250) — matrix operand order repaired and checked by reading, not compiled, not run on hardware
//
#include <hip/hip_runtime.h>
#include <stdint.h>

// ---------------------------------------------------------------------------
// GraphMamba pipeline for MI455X (gfx1250, wave32, WMMA)
//   GCN GEMMs  -> v_wmma_f32_16x16x32_f16, f16 staged in LDS via b128 stores,
//                 f32 accumulate; inter-layer activations kept in f16
//   GCN scatter-> global_atomic_add_f32 (aggregation buffer is L2-resident)
//   LSTM/head  -> small VALU kernels (serial / negligible)
// ---------------------------------------------------------------------------

typedef __attribute__((ext_vector_type(16))) _Float16 v16h;
typedef __attribute__((ext_vector_type(8)))  float    v8f;

union Frag16 { v16h v; _Float16 h[16]; uint32_t u[8]; };
union FragF  { v8f  v; float f[8]; };

#define NCH 128   // feature width D

// pack two floats into two RNE-converted halves in one dword
__device__ __forceinline__ uint32_t pkh2(float a, float b) {
    union { _Float16 h[2]; uint32_t u; } p;
    p.h[0] = (_Float16)a; p.h[1] = (_Float16)b;
    return p.u;
}

// ---------------------------------------------------------------------------
// GEMM: C[M x 128] = A[M x K] * Bw[K x 128], K in {32,128}, M % 128 == 0.
// 256 threads = 8 waves; each wave owns a 16-row strip, all 128 columns
// (8 WMMA 16x16 accumulator tiles). A and B^T staged in LDS as f16 with
// packed b128 stores. T = float (layer 1 input) or _Float16 (f16 acts).
// Fragment element mapping follows CDNA5 ISA 16-bit layouts.
// ---------------------------------------------------------------------------
template <typename T>
__global__ void __launch_bounds__(256)
gemm_wmma_kernel(const T* __restrict__ A, const float* __restrict__ Bw,
                 float* __restrict__ C, int K)
{
    __shared__ __align__(16) _Float16 sA[128 * 32];   // [row][k]
    __shared__ __align__(16) _Float16 sBt[128 * 32];  // [n][k] (transposed B)

    const int tid  = threadIdx.x;
    const int lane = tid & 31;
    const int wave = tid >> 5;
    const int hi   = lane >> 4;
    const int l15  = lane & 15;
    const long m0  = (long)blockIdx.x * 128;

    // staging coordinates: each thread fills half an A row (16 halves) and a
    // half B^T row (16 halves) with a single pair of b128 LDS stores.
    const int ar  = tid >> 1;              // A row within tile
    const int ac0 = (tid & 1) * 16;        // A k-offset within tile
    const int bn  = tid & 127;             // B^T n
    const int bc0 = (tid >> 7) * 16;       // B^T k-offset

    FragF acc[8];
    #pragma unroll
    for (int j = 0; j < 8; ++j)
        #pragma unroll
        for (int r = 0; r < 8; ++r) acc[j].f[r] = 0.0f;

    for (int k0 = 0; k0 < K; k0 += 32) {
        // ---- stage A tile (vectorized) ----
        const T* ap = A + (m0 + ar) * (long)K + (k0 + ac0);
        uint4 a0, a1;
        if constexpr (sizeof(T) == 4) {            // f32 -> f16 convert+pack
            float4 f0 = *(const float4*)(ap + 0);
            float4 f1 = *(const float4*)(ap + 4);
            float4 f2 = *(const float4*)(ap + 8);
            float4 f3 = *(const float4*)(ap + 12);
            a0 = uint4{pkh2(f0.x, f0.y), pkh2(f0.z, f0.w),
                       pkh2(f1.x, f1.y), pkh2(f1.z, f1.w)};
            a1 = uint4{pkh2(f2.x, f2.y), pkh2(f2.z, f2.w),
                       pkh2(f3.x, f3.y), pkh2(f3.z, f3.w)};
        } else {                                   // already f16: b128 copies
            a0 = *(const uint4*)(ap + 0);
            a1 = *(const uint4*)(ap + 8);
        }
        *(uint4*)&sA[tid * 16 + 0] = a0;
        *(uint4*)&sA[tid * 16 + 8] = a1;

        // ---- stage B^T tile (coalesced reads across lanes, b128 stores) ----
        uint32_t uu[8];
        #pragma unroll
        for (int i = 0; i < 8; ++i)
            uu[i] = pkh2(Bw[(k0 + bc0 + 2 * i) * NCH + bn],
                         Bw[(k0 + bc0 + 2 * i + 1) * NCH + bn]);
        *(uint4*)&sBt[bn * 32 + bc0 + 0] = uint4{uu[0], uu[1], uu[2], uu[3]};
        *(uint4*)&sBt[bn * 32 + bc0 + 8] = uint4{uu[4], uu[5], uu[6], uu[7]};

        // prefetch next A slab while this tile computes (global_prefetch_b8)
        if (k0 + 32 < K) __builtin_prefetch(ap + 32, 0, 0);

        __syncthreads();

        // A fragment: two packed b128 LDS reads per lane
        Frag16 a;
        #pragma unroll
        for (int vv = 0; vv < 8; ++vv) {
            int kk = ((vv < 4) ? 0 : 16) + (hi ? 8 : 0) + 2 * (vv & 3);
            a.u[vv] = *(const uint32_t*)&sA[(wave * 16 + l15) * 32 + kk];
        }
        #pragma unroll
        for (int j = 0; j < 8; ++j) {
            Frag16 b;
            const int kb = hi ? 16 : 0;
            #pragma unroll
            for (int vv = 0; vv < 8; ++vv)
                b.u[vv] = *(const uint32_t*)&sBt[(j * 16 + l15) * 32 + ((vv < 4) ? 0 : 16) + (hi ? 8 : 0) + 2 * (vv & 3)];
            acc[j].v = __builtin_amdgcn_wmma_f32_16x16x32_f16(
                false, a.v, false, b.v, (short)0, acc[j].v, false, false);
        }
        __syncthreads();
    }

    // Store: C/D layout => vgpr r: lanes0-15 row m0+wave*16+r, lanes16-31 +8
    #pragma unroll
    for (int j = 0; j < 8; ++j) {
        #pragma unroll
        for (int r = 0; r < 8; ++r) {
            long row = m0 + wave * 16 + r + (hi ? 8 : 0);
            C[row * NCH + (j * 16 + l15)] = acc[j].f[r];
        }
    }
}

// ---------------------------------------------------------------------------
// Degree / normalization
// ---------------------------------------------------------------------------
__global__ void deg_kernel(const int* __restrict__ ei, float* __restrict__ deg,
                           int E, int N)
{
    int e = blockIdx.x * blockDim.x + threadIdx.x;
    if (e >= E + N) return;
    int t = (e < E) ? ei[E + e] : (e - E);   // dst, with self loops appended
    atomicAdd(&deg[t], 1.0f);
}

__global__ void dinv_kernel(const float* __restrict__ deg, float* __restrict__ dinv)
{
    int i = blockIdx.x * blockDim.x + threadIdx.x;
    if (i < 1024) dinv[i] = rsqrtf(deg[i]);
}

// ---------------------------------------------------------------------------
// GCN message scatter: out[g, dst, :] += H[g, src, :] * dinv[src]*dinv[dst]
// One block per (edge, graph); atomics resolve in L2 (32MB buffer << 192MB L2)
// ---------------------------------------------------------------------------
__global__ void scatter_kernel(const float* __restrict__ H, const int* __restrict__ ei,
                               const float* __restrict__ dinv, float* __restrict__ out,
                               int E, int N)
{
    int e = blockIdx.x;
    int g = blockIdx.y;
    int d = threadIdx.x;
    int s, t;
    if (e < E) { s = ei[e]; t = ei[E + e]; } else { s = e - E; t = s; }
    float nm = dinv[s] * dinv[t];
    const float* hp = H   + ((long)g * N + s) * NCH;
    float*       op = out + ((long)g * N + t) * NCH;
    atomicAdd(&op[d], hp[d] * nm);
}

// ---------------------------------------------------------------------------
// out(f16) = silu(in + bias); 4 elements/thread: float4 load, uint2 store
// ---------------------------------------------------------------------------
__global__ void bias_silu_kernel(const float* __restrict__ in,
                                 const float* __restrict__ bias,
                                 _Float16* __restrict__ out, long total4)
{
    long q = (long)blockIdx.x * blockDim.x + threadIdx.x;
    if (q >= total4) return;
    long base = q * 4;
    float4 v = *(const float4*)&in[base];
    int bidx = (int)(base & (NCH - 1));   // 4 | 128, no wrap within the quad
    v.x += bias[bidx + 0]; v.y += bias[bidx + 1];
    v.z += bias[bidx + 2]; v.w += bias[bidx + 3];
    float s0 = v.x / (1.0f + __expf(-v.x));
    float s1 = v.y / (1.0f + __expf(-v.y));
    float s2 = v.z / (1.0f + __expf(-v.z));
    float s3 = v.w / (1.0f + __expf(-v.w));
    *(uint2*)&out[base] = uint2{pkh2(s0, s1), pkh2(s2, s3)};
}

// ---------------------------------------------------------------------------
// emb[g, d] = mean over N nodes of P[g, n, d]   (P is f16, f32 accumulate)
// ---------------------------------------------------------------------------
__global__ void mean_kernel(const _Float16* __restrict__ P, float* __restrict__ emb,
                            int N)
{
    int g = blockIdx.x, d = threadIdx.x;
    const _Float16* p = P + (long)g * N * NCH + d;
    float s = 0.0f;
    for (int n = 0; n < N; ++n) s += (float)p[(long)n * NCH];
    emb[g * NCH + d] = s * (1.0f / (float)N);
}

// ---------------------------------------------------------------------------
// One LSTM layer, one block per batch element (independent), 512 threads =
// one thread per gate row. h/c/x/gates in LDS; serial over T steps.
// ---------------------------------------------------------------------------
__global__ void __launch_bounds__(512)
lstm_kernel(const float* __restrict__ xseq, const float* __restrict__ Wih,
            const float* __restrict__ Whh, const float* __restrict__ bih,
            const float* __restrict__ bhh, float* __restrict__ hseq, int T)
{
    __shared__ float xs[NCH], hs[NCH], cs[NCH], gs[4 * NCH];
    const int b = blockIdx.x;
    const int g = threadIdx.x;          // 0..511
    if (g < NCH) { hs[g] = 0.0f; cs[g] = 0.0f; }
    __syncthreads();

    for (int t = 0; t < T; ++t) {
        if (g < NCH) xs[g] = xseq[((long)b * T + t) * NCH + g];
        __syncthreads();

        float acc = bih[g] + bhh[g];
        const float* wi = Wih + (long)g * NCH;
        const float* wh = Whh + (long)g * NCH;
        #pragma unroll 4
        for (int k = 0; k < NCH; ++k) acc += wi[k] * xs[k] + wh[k] * hs[k];
        gs[g] = acc;
        __syncthreads();

        if (g < NCH) {
            float ig = 1.0f / (1.0f + __expf(-gs[g]));
            float fg = 1.0f / (1.0f + __expf(-gs[NCH + g]));
            float gg = tanhf(gs[2 * NCH + g]);
            float og = 1.0f / (1.0f + __expf(-gs[3 * NCH + g]));
            float c  = fg * cs[g] + ig * gg;
            float h  = og * tanhf(c);
            cs[g] = c; hs[g] = h;
            hseq[((long)b * T + t) * NCH + g] = h;
        }
        __syncthreads();
    }
}

// ---------------------------------------------------------------------------
// Head: out = silu(h_final @ Wh1 + bh1) @ Wh2 + bh2 ; 2 batches x 64 hidden
// ---------------------------------------------------------------------------
__global__ void head_kernel(const float* __restrict__ hseq,
                            const float* __restrict__ Wh1, const float* __restrict__ bh1,
                            const float* __restrict__ Wh2, const float* __restrict__ bh2,
                            float* __restrict__ out, int T)
{
    __shared__ float hh[2][64];
    int b = threadIdx.x >> 6;           // blockDim = 128
    int j = threadIdx.x & 63;
    const float* fs = hseq + ((long)b * T + (T - 1)) * NCH;
    float a = bh1[j];
    for (int k = 0; k < NCH; ++k) a += fs[k] * Wh1[k * 64 + j];
    hh[b][j] = a / (1.0f + __expf(-a));
    __syncthreads();
    if (j < 2) {
        float o = bh2[j];
        for (int k = 0; k < 64; ++k) o += hh[b][k] * Wh2[k * 2 + j];
        out[b * 2 + j] = o;
    }
}

// ---------------------------------------------------------------------------
extern "C" void kernel_launch(void* const* d_in, const int* in_sizes, int n_in,
                              void* d_out, int out_size, void* d_ws, size_t ws_size,
                              hipStream_t stream)
{
    const float* X    = (const float*)d_in[0];
    const int*   ei   = (const int*)  d_in[1];   // (2, E) int32
    const float* W1   = (const float*)d_in[2];
    const float* b1   = (const float*)d_in[3];
    const float* W2   = (const float*)d_in[4];
    const float* b2   = (const float*)d_in[5];
    const float* W3   = (const float*)d_in[6];
    const float* b3   = (const float*)d_in[7];
    const float* Wih0 = (const float*)d_in[8];
    const float* Whh0 = (const float*)d_in[9];
    const float* bih0 = (const float*)d_in[10];
    const float* bhh0 = (const float*)d_in[11];
    const float* Wih1 = (const float*)d_in[12];
    const float* Whh1 = (const float*)d_in[13];
    const float* bih1 = (const float*)d_in[14];
    const float* bhh1 = (const float*)d_in[15];
    const float* Wh1  = (const float*)d_in[16];
    const float* bh1  = (const float*)d_in[17];
    const float* Wh2  = (const float*)d_in[18];
    const float* bh2  = (const float*)d_in[19];

    const int  Bx = 2, T = 32, N = 1024, E = 16384;
    const long GT = (long)Bx * T;        // 64 graph snapshots
    const long MN = GT * N;              // 65536 node rows
    const long f32Bytes = MN * NCH * sizeof(float);     // 32 MB
    const long f16Bytes = MN * NCH * sizeof(_Float16);  // 16 MB

    char* ws = (char*)d_ws;
    float*    G    = (float*)ws;     ws += f32Bytes;    // GEMM output
    float*    Agg  = (float*)ws;     ws += f32Bytes;    // aggregation buffer
    _Float16* P    = (_Float16*)ws;  ws += f16Bytes;    // f16 activations
    float*    deg  = (float*)ws;     ws += 1024 * sizeof(float);
    float*    dinv = (float*)ws;     ws += 1024 * sizeof(float);
    float*    emb  = (float*)ws;     ws += GT * NCH * sizeof(float);
    float*    hs0  = (float*)ws;     ws += GT * NCH * sizeof(float);
    float*    hs1  = (float*)ws;     ws += GT * NCH * sizeof(float);

    // degree normalization (with self loops)
    hipMemsetAsync(deg, 0, 1024 * sizeof(float), stream);
    deg_kernel<<<(E + N + 255) / 256, 256, 0, stream>>>(ei, deg, E, N);
    dinv_kernel<<<4, 256, 0, stream>>>(deg, dinv);

    const dim3 sgrid(E + N, (unsigned)GT);
    const long total4 = MN * NCH / 4;
    const unsigned sil_blocks = (unsigned)((total4 + 255) / 256);
    const unsigned gemm_blocks = (unsigned)(MN / 128);

    // GCN layer 1 (K = F = 32, f32 input)
    gemm_wmma_kernel<float><<<gemm_blocks, 256, 0, stream>>>(X, W1, G, 32);
    hipMemsetAsync(Agg, 0, f32Bytes, stream);
    scatter_kernel<<<sgrid, NCH, 0, stream>>>(G, ei, dinv, Agg, E, N);
    bias_silu_kernel<<<sil_blocks, 256, 0, stream>>>(Agg, b1, P, total4);

    // GCN layer 2 (K = 128, f16 input)
    gemm_wmma_kernel<_Float16><<<gemm_blocks, 256, 0, stream>>>(P, W2, G, 128);
    hipMemsetAsync(Agg, 0, f32Bytes, stream);
    scatter_kernel<<<sgrid, NCH, 0, stream>>>(G, ei, dinv, Agg, E, N);
    bias_silu_kernel<<<sil_blocks, 256, 0, stream>>>(Agg, b2, P, total4);

    // GCN layer 3 (K = 128, f16 input)
    gemm_wmma_kernel<_Float16><<<gemm_blocks, 256, 0, stream>>>(P, W3, G, 128);
    hipMemsetAsync(Agg, 0, f32Bytes, stream);
    scatter_kernel<<<sgrid, NCH, 0, stream>>>(G, ei, dinv, Agg, E, N);
    bias_silu_kernel<<<sil_blocks, 256, 0, stream>>>(Agg, b3, P, total4);

    // node mean -> (B*T, 128)
    mean_kernel<<<(unsigned)GT, NCH, 0, stream>>>(P, emb, N);

    // two LSTM layers (per-batch blocks, serial over T)
    lstm_kernel<<<Bx, 512, 0, stream>>>(emb, Wih0, Whh0, bih0, bhh0, hs0, T);
    lstm_kernel<<<Bx, 512, 0, stream>>>(hs0, Wih1, Whh1, bih1, bhh1, hs1, T);

    // head
    head_kernel<<<1, 128, 0, stream>>>(hs1, Wh1, bh1, Wh2, bh2, (float*)d_out, T);
}
